// _LSTMModel_58428735095035
// MI455X (gfx1250) — matrix operand order repaired and checked by reading, not compiled, not run on hardware
//
#include <hip/hip_runtime.h>

typedef _Float16 h8   __attribute__((ext_vector_type(8)));
typedef _Float16 v16h __attribute__((ext_vector_type(16)));
typedef float    v8f  __attribute__((ext_vector_type(8)));

#define B_  4096
#define T_  256
#define F_  32
#define H1_ 64
#define H2_ 32

// gfx1250 has a native v_tanh_f32 TRANS op; sigmoid via tanh identity (no divides).
#if __has_builtin(__builtin_amdgcn_tanhf)
__device__ __forceinline__ float tanh_fast(float x) { return __builtin_amdgcn_tanhf(x); }
#else
__device__ __forceinline__ float tanh_fast(float x) {
  const float e = __expf(-2.0f * x);
  return (1.0f - e) * __builtin_amdgcn_rcpf(1.0f + e);
}
#endif
__device__ __forceinline__ float sig_fast(float x) { return fmaf(0.5f, tanh_fast(0.5f * x), 0.5f); }

__device__ __forceinline__ v16h mk16(h8 lo, h8 hi) {
  return __builtin_shufflevector(lo, hi, 0, 1, 2, 3, 4, 5, 6, 7, 8, 9, 10, 11, 12, 13, 14, 15);
}

// A-fragment (16 x 32 K-chunk) from row-major f16 LDS, row stride ld (halves).
// ISA 7.12.2 16-bit A layout: lanes 0-15 rows; halves[0..7]=K hf*8.., halves[8..15]=K 16+hf*8..
__device__ __forceinline__ v16h ldsA(const _Float16* __restrict__ base, int ld, int lane) {
  const int m = lane & 15, hf = lane >> 4;
  const _Float16* p = base + m * ld + hf * 8;
  h8 lo = *(const h8*)(p);
  h8 hi = *(const h8*)(p + 16);
  return mk16(lo, hi);
}

// B-fragment for gates = A @ W^T : B[k][n] = W[n][k], W row-major [N][ldk] fp32 in GLOBAL.
__device__ __forceinline__ v16h globB(const float* __restrict__ W, int ldk, int n0, int k0, int lane) {
  const int n  = n0 + (lane & 15);
  const int kk = k0 + ((lane >> 4) << 3);
  const float* p = W + n * ldk + kk;
  float f[16];
  *(float4*)(f + 0)  = *(const float4*)(p + 0);
  *(float4*)(f + 4)  = *(const float4*)(p + 4);
  *(float4*)(f + 8)  = *(const float4*)(p + 16);
  *(float4*)(f + 12) = *(const float4*)(p + 20);
  v16h r;
#pragma unroll
  for (int i = 0; i < 16; ++i) r[i] = (_Float16)f[i];
  return r;
}

__device__ __forceinline__ v8f wmma16(v16h a, v16h b, v8f c) {
  return __builtin_amdgcn_wmma_f32_16x16x32_f16(false, a, false, b, (short)0, c, false, false);
}

// 6 waves: waves 0-3 = layer-1 pipeline stage (step t), waves 4-5 = layer-2 stage (step t-1).
__global__ void __launch_bounds__(192)
lstm2_fused_kernel(const float* __restrict__ x,
                   const float* __restrict__ Wih1, const float* __restrict__ Whh1,
                   const float* __restrict__ bih1, const float* __restrict__ bhh1,
                   const float* __restrict__ Wih2, const float* __restrict__ Whh2,
                   const float* __restrict__ bih2, const float* __restrict__ bhh2,
                   const float* __restrict__ Wfc1, const float* __restrict__ bfc1,
                   const float* __restrict__ Wfc2, const float* __restrict__ bfc2,
                   float* __restrict__ out) {
  __shared__ __align__(16) _Float16 sH1[2 * 16 * H1_];  // h1(t) lives in buf[t&1]
  __shared__ __align__(16) _Float16 sH2[2 * 16 * H2_];  // h2(s) lives in buf[s&1]
  __shared__ __align__(16) float    sHlast[16 * H2_];   // final h2, fp32

  const int tid  = threadIdx.x;
  const int lane = tid & 31;
  const int wave = tid >> 5;
  const int ln   = lane & 15;
  const int mhi  = (lane >> 4) << 3;  // C-layout: lanes 16-31 hold rows 8..15
  const int m0   = blockIdx.x * 16;
  const bool isL1 = (wave < 4);

  for (int i = tid; i < 2 * 16 * H1_; i += 192) sH1[i] = (_Float16)0.0f;
  for (int i = tid; i < 2 * 16 * H2_; i += 192) sH2[i] = (_Float16)0.0f;

  // Per-wave weight fragments (only this wave's pipeline stage):
  //  L1 wave w (cols [w*16,w*16+16)): Bx0 = Wih1 tile, Bk0/Bk1 = Whh1 K-chunks, tiles n0=G*64+w*16
  //  L2 wave w-4 (cols [(w-4)*16,..)): Bk0/Bk1 = Wih2 K-chunks, Bx0 = Whh2 tile, tiles n0=G*32+(w-4)*16
  v16h Bx0[4], Bk0[4], Bk1[4];
  float biasv[4];
  if (isL1) {
#pragma unroll
    for (int G = 0; G < 4; ++G) {
      const int n1 = G * 64 + wave * 16;
      Bx0[G] = globB(Wih1, F_,  n1, 0,  lane);
      Bk0[G] = globB(Whh1, H1_, n1, 0,  lane);
      Bk1[G] = globB(Whh1, H1_, n1, 32, lane);
      biasv[G] = bih1[n1 + ln] + bhh1[n1 + ln];
    }
  } else {
    const int w2 = wave - 4;
#pragma unroll
    for (int G = 0; G < 4; ++G) {
      const int n2 = G * 32 + w2 * 16;
      Bk0[G] = globB(Wih2, H1_, n2, 0,  lane);
      Bk1[G] = globB(Wih2, H1_, n2, 32, lane);
      Bx0[G] = globB(Whh2, H2_, n2, 0,  lane);
      biasv[G] = bih2[n2 + ln] + bhh2[n2 + ln];
    }
  }

  v8f cst = {};  // c-state for this wave's columns (c1 for L1 waves, c2 for L2 waves)
  const v8f zro = {};

  __syncthreads();

  const float* xrow = x + ((size_t)(m0 + ln) * T_) * F_ + (lane >> 4) * 8;

  for (int it = 0; it <= T_; ++it) {
    const int pA = (it + 1) & 1;  // parity of it-1
    const int pB = it & 1;

    if (isL1) {
      if (it < T_) {
        const _Float16* rH1 = sH1 + pA * (16 * H1_);  // h1(it-1)
        _Float16*       wH1 = sH1 + pB * (16 * H1_);  // h1(it)

        const float* q = xrow + (size_t)it * F_;
        if (it + 8 < T_) __builtin_prefetch(q + 8 * F_, 0, 1);
        float fx[16];
        *(float4*)(fx + 0)  = *(const float4*)(q + 0);
        *(float4*)(fx + 4)  = *(const float4*)(q + 4);
        *(float4*)(fx + 8)  = *(const float4*)(q + 16);
        *(float4*)(fx + 12) = *(const float4*)(q + 20);
        v16h a1;
#pragma unroll
        for (int i = 0; i < 16; ++i) a1[i] = (_Float16)fx[i];

        v16h ah0 = ldsA(rH1, H1_, lane);
        v16h ah1 = ldsA(rH1 + 32, H1_, lane);

        v8f acc[4];  // G = 0:i 1:f 2:g 3:o
#pragma unroll
        for (int G = 0; G < 4; ++G) {
          acc[G] = wmma16(a1,  Bx0[G], zro);
          acc[G] = wmma16(ah0, Bk0[G], acc[G]);
          acc[G] = wmma16(ah1, Bk1[G], acc[G]);
        }

        const int j = wave * 16 + ln;
#pragma unroll
        for (int r = 0; r < 8; ++r) {
          const float iv = sig_fast(acc[0][r] + biasv[0]);
          const float fv = sig_fast(acc[1][r] + biasv[1]);
          const float gv = tanh_fast(acc[2][r] + biasv[2]);
          const float ov = sig_fast(acc[3][r] + biasv[3]);
          cst[r] = fv * cst[r] + iv * gv;
          const float hv = ov * tanh_fast(cst[r]);
          wH1[(mhi + r) * H1_ + j] = (_Float16)hv;
        }
      }
    } else {
      if (it > 0) {
        const int s = it - 1;
        const _Float16* rH1 = sH1 + pA * (16 * H1_);  // h1(s)
        const _Float16* rH2 = sH2 + pB * (16 * H2_);  // h2(s-1)
        _Float16*       wH2 = sH2 + pA * (16 * H2_);  // h2(s)

        v16h a20 = ldsA(rH1, H1_, lane);
        v16h a21 = ldsA(rH1 + 32, H1_, lane);
        v16h ah2 = ldsA(rH2, H2_, lane);

        v8f acc[4];
#pragma unroll
        for (int G = 0; G < 4; ++G) {
          acc[G] = wmma16(a20, Bk0[G], zro);
          acc[G] = wmma16(a21, Bk1[G], acc[G]);
          acc[G] = wmma16(ah2, Bx0[G], acc[G]);
        }

        const int j = (wave - 4) * 16 + ln;
#pragma unroll
        for (int r = 0; r < 8; ++r) {
          const float iv = sig_fast(acc[0][r] + biasv[0]);
          const float fv = sig_fast(acc[1][r] + biasv[1]);
          const float gv = tanh_fast(acc[2][r] + biasv[2]);
          const float ov = sig_fast(acc[3][r] + biasv[3]);
          cst[r] = fv * cst[r] + iv * gv;
          const float hv = ov * tanh_fast(cst[r]);
          wH2[(mhi + r) * H2_ + j] = (_Float16)hv;
          if (s == T_ - 1) sHlast[(mhi + r) * H2_ + j] = hv;
        }
      }
    }
    __syncthreads();  // one barrier per pipelined iteration
  }

  // ---------------- head: out = relu(h_last @ Wfc1^T + b1) @ Wfc2^T + b2 ----------
  if (tid < 16) {
    const int m = tid;
    float o2 = bfc2[0];
    for (int d = 0; d < 16; ++d) {
      float a = bfc1[d];
      for (int k = 0; k < 32; ++k) a += sHlast[m * 32 + k] * Wfc1[d * 32 + k];
      a = fmaxf(a, 0.0f);
      o2 += a * Wfc2[d];
    }
    out[m0 + m] = o2;
  }
}

extern "C" void kernel_launch(void* const* d_in, const int* in_sizes, int n_in,
                              void* d_out, int out_size, void* d_ws, size_t ws_size,
                              hipStream_t stream) {
  (void)in_sizes; (void)n_in; (void)out_size; (void)d_ws; (void)ws_size;
  const float* x    = (const float*)d_in[0];
  const float* Wih1 = (const float*)d_in[1];
  const float* Whh1 = (const float*)d_in[2];
  const float* bih1 = (const float*)d_in[3];
  const float* bhh1 = (const float*)d_in[4];
  const float* Wih2 = (const float*)d_in[5];
  const float* Whh2 = (const float*)d_in[6];
  const float* bih2 = (const float*)d_in[7];
  const float* bhh2 = (const float*)d_in[8];
  const float* Wfc1 = (const float*)d_in[9];
  const float* bfc1 = (const float*)d_in[10];
  const float* Wfc2 = (const float*)d_in[11];
  const float* bfc2 = (const float*)d_in[12];
  float* out = (float*)d_out;

  dim3 grid(B_ / 16), block(192);
  lstm2_fused_kernel<<<grid, block, 0, stream>>>(x, Wih1, Whh1, bih1, bhh1,
                                                 Wih2, Whh2, bih2, bhh2,
                                                 Wfc1, bfc1, Wfc2, bfc2, out);
}
